// EdgeGNN_1176821039617
// MI455X (gfx1250) — matrix operand order repaired and checked by reading, not compiled, not run on hardware
//
#include <hip/hip_runtime.h>
#include <hip/hip_bf16.h>

typedef __attribute__((ext_vector_type(16))) _Float16 v16h;
typedef __attribute__((ext_vector_type(8)))  _Float16 v8h;
typedef __attribute__((ext_vector_type(8)))  float    v8f;
typedef __attribute__((ext_vector_type(4)))  float    v4f;

#define N_NODES 100000
#define N_EDGES 3200000
#define NODE_IN 128
#define EDGE_IN 16
#define HIDDEN  32
#define MLP_H   64

#define NODE_TILES (N_NODES / 16)   // 6250
#define EDGE_TILES (N_EDGES / 16)   // 200000

// ---------------------------------------------------------------------------
// Kernel 1: per-node encoder  x_enc[n, 0:32] = f16( x[n,:] @ W_enc + b_enc )
// One wave = 16-node M-tiles (grid-stride).  M=16, N=32 (2 tiles), K=128
// (4 chunks of 32).  W_enc is staged once into LDS in B-fragment layout,
// then hoisted into 8 register-resident v16h fragments per wave.
// ---------------------------------------------------------------------------
__global__ __launch_bounds__(256) void node_encode_kernel(
    const float* __restrict__ x,
    const float* __restrict__ W_enc,
    const float* __restrict__ b_enc,
    _Float16*    __restrict__ x_enc)
{
    // B fragments: [t(2)][c(4)][lane(32)][elem(16)] f16  = 8 KB
    __shared__ __align__(32) _Float16 sB[2 * 4 * 32 * 16];
    __shared__ float sb[HIDDEN];

    // Pre-swizzle W_enc (k-major [128][32]) into B-fragment order.
    // B layout (16-bit 32x16): lane L -> n = L%16 ; k = 8*(L/16) + (elem%8) + 16*(elem/8).
    for (int fi = (int)threadIdx.x; fi < 2 * 4 * 32 * 16; fi += 256) {
        const int i = fi & 15;
        const int L = (fi >> 4) & 31;
        const int c = (fi >> 9) & 3;
        const int t = (fi >> 11) & 1;
        const int k = c * 32 + ((L >> 4) << 3) + (i & 7) + ((i >> 3) << 4);     // 0..127
        const int n = t * 16 + (L & 15);                // 0..31
        sB[fi] = (_Float16)W_enc[k * HIDDEN + n];
    }
    if (threadIdx.x < HIDDEN) sb[threadIdx.x] = b_enc[threadIdx.x];
    __syncthreads();

    const int wave  = threadIdx.x >> 5;
    const int lane  = threadIdx.x & 31;
    const int m     = lane & 15;
    const int khalf = lane >> 4;

    // Hoist the 8 B fragments (2 N-tiles x 4 K-chunks) into registers.
    v16h B[8];
#pragma unroll
    for (int f = 0; f < 8; ++f)
        B[f] = *(const v16h*)&sB[(f * 32 + lane) * 16];

    const float bj0 = sb[m];
    const float bj1 = sb[16 + m];

    const int gwave  = blockIdx.x * 8 + wave;
    const int stride = gridDim.x * 8;

    for (int tile = gwave; tile < NODE_TILES; tile += stride) {
        const int node = tile * 16 + m;
        const float* __restrict__ xrow = x + (long)node * NODE_IN;

        v8f acc0 = {}; v8f acc1 = {};

#pragma unroll
        for (int c = 0; c < 4; ++c) {
            // A fragment (16-bit A 16x32):
            // lane<16 holds K {0-7,16-23}, lane>=16 holds K {8-15,24-31}
            v16h a;
            const int k0 = c * 32 + khalf * 8;
            const int k1 = c * 32 + 16 + khalf * 8;
            const v4f g0a = *(const v4f*)(xrow + k0);
            const v4f g0b = *(const v4f*)(xrow + k0 + 4);
            const v4f g1a = *(const v4f*)(xrow + k1);
            const v4f g1b = *(const v4f*)(xrow + k1 + 4);
#pragma unroll
            for (int i = 0; i < 4; ++i) {
                a[i]      = (_Float16)g0a[i];
                a[4 + i]  = (_Float16)g0b[i];
                a[8 + i]  = (_Float16)g1a[i];
                a[12 + i] = (_Float16)g1b[i];
            }
            acc0 = __builtin_amdgcn_wmma_f32_16x16x32_f16(false, a, false, B[0 * 4 + c],
                                                          (short)0, acc0, false, false);
            acc1 = __builtin_amdgcn_wmma_f32_16x16x32_f16(false, a, false, B[1 * 4 + c],
                                                          (short)0, acc1, false, false);
        }

        // C layout: lane holds column n = t*16 + m ; VGPR r holds row r + 8*khalf.
#pragma unroll
        for (int r = 0; r < 8; ++r) {
            const int row = r + 8 * khalf;
            const long base = (long)(tile * 16 + row) * HIDDEN;
            x_enc[base + m]      = (_Float16)(acc0[r] + bj0);
            x_enc[base + 16 + m] = (_Float16)(acc1[r] + bj1);
        }
    }
}

// ---------------------------------------------------------------------------
// Kernel 2: per-edge MLP.  One wave = 16-edge tiles (grid-stride).
//   A (16x96 f16) = [x_enc[src](32) | x_enc[dst](32) | edge_attr(16)+pad(16)]
//   layer1: 4 N-tiles x 3 K-chunks of v_wmma_f32_16x16x32_f16, B fragments
//           register-resident (loaded once from LDS after one stage+barrier)
//   layer2: relu(acc+b1)*W2 folded into accumulator layout + shfl_xor reduce
// ---------------------------------------------------------------------------
__global__ __launch_bounds__(256) void edge_mlp_kernel(
    const _Float16*  __restrict__ x_enc,
    const long long* __restrict__ edge_index,
    const float*     __restrict__ edge_attr,
    const float*     __restrict__ W1,
    const float*     __restrict__ b1,
    const float*     __restrict__ W2,
    const float*     __restrict__ b2,
    float*           __restrict__ out)
{
    // B fragments: [t(4)][c(3)][lane(32)][elem(16)] f16 = 12 KB (K padded 80->96)
    __shared__ __align__(32) _Float16 sB[4 * 3 * 32 * 16];
    __shared__ float sb1[MLP_H];
    __shared__ float sW2[MLP_H];

    for (int fi = (int)threadIdx.x; fi < 4 * 3 * 32 * 16; fi += 256) {
        const int i = fi & 15;
        const int L = (fi >> 4) & 31;
        const int g = fi >> 9;            // 0..11
        const int c = g % 3;
        const int t = g / 3;
        const int k = c * 32 + ((L >> 4) << 3) + (i & 7) + ((i >> 3) << 4);     // 0..95
        const int n = t * 16 + (L & 15);                // 0..63
        sB[fi] = (k < 2 * HIDDEN + EDGE_IN) ? (_Float16)W1[k * MLP_H + n]
                                            : (_Float16)0.f;
    }
    if (threadIdx.x < MLP_H) {
        sb1[threadIdx.x] = b1[threadIdx.x];
        sW2[threadIdx.x] = W2[threadIdx.x];
    }
    __syncthreads();

    const int wave  = threadIdx.x >> 5;
    const int lane  = threadIdx.x & 31;
    const int m     = lane & 15;
    const int khalf = lane >> 4;

    // Hoist all 12 layer-1 B fragments into registers (96 VGPRs), reused
    // across every edge tile this wave processes.
    v16h B[12];
#pragma unroll
    for (int f = 0; f < 12; ++f)
        B[f] = *(const v16h*)&sB[(f * 32 + lane) * 16];

    // Per-lane layer-2 constants for the 4 N-tiles (column n = t*16 + m).
    float bj[4], wj[4];
#pragma unroll
    for (int t = 0; t < 4; ++t) {
        bj[t] = sb1[t * 16 + m];
        wj[t] = sW2[t * 16 + m];
    }
    const float b2v = b2[0];

    const long gwave  = (long)blockIdx.x * 8 + wave;
    const long stride = (long)gridDim.x * 8;

    for (long tile = gwave; tile < EDGE_TILES; tile += stride) {
        const long e = tile * 16 + m;
        const int src = (int)edge_index[e];
        const int dst = (int)edge_index[(long)N_EDGES + e];

        // --- A fragments (gathers hit the L2-resident 6.4 MB f16 table) ---
        v16h a0, a1, a2;
        {
            const _Float16* __restrict__ ps = x_enc + (long)src * HIDDEN;
            const v8h g0 = *(const v8h*)(ps + khalf * 8);
            const v8h g1 = *(const v8h*)(ps + 16 + khalf * 8);
#pragma unroll
            for (int i = 0; i < 8; ++i) { a0[i] = g0[i]; a0[8 + i] = g1[i]; }
        }
        {
            const _Float16* __restrict__ pd = x_enc + (long)dst * HIDDEN;
            const v8h g0 = *(const v8h*)(pd + khalf * 8);
            const v8h g1 = *(const v8h*)(pd + 16 + khalf * 8);
#pragma unroll
            for (int i = 0; i < 8; ++i) { a1[i] = g0[i]; a1[8 + i] = g1[i]; }
        }
        {
            // chunk-local K 0..15 = edge_attr, K 16..31 = zero pad
            const float* __restrict__ pa = edge_attr + e * EDGE_IN + khalf * 8;
            const v4f q0 = *(const v4f*)pa;
            const v4f q1 = *(const v4f*)(pa + 4);
#pragma unroll
            for (int i = 0; i < 4; ++i) {
                a2[i]     = (_Float16)q0[i];
                a2[4 + i] = (_Float16)q1[i];
            }
#pragma unroll
            for (int i = 8; i < 16; ++i) a2[i] = (_Float16)0.f;
        }

        v8f partial = {};

#pragma unroll
        for (int t = 0; t < 4; ++t) {
            v8f acc = {};
            acc = __builtin_amdgcn_wmma_f32_16x16x32_f16(false, a0, false, B[t * 3 + 0],
                                                         (short)0, acc, false, false);
            acc = __builtin_amdgcn_wmma_f32_16x16x32_f16(false, a1, false, B[t * 3 + 1],
                                                         (short)0, acc, false, false);
            acc = __builtin_amdgcn_wmma_f32_16x16x32_f16(false, a2, false, B[t * 3 + 2],
                                                         (short)0, acc, false, false);
            // lane owns column n = t*16 + m of h; fold bias+relu+W2 here
#pragma unroll
            for (int r = 0; r < 8; ++r) {
                float h = acc[r] + bj[t];
                h = (h > 0.f) ? h : 0.f;
                partial[r] += h * wj[t];
            }
        }

        // Reduce the 64 columns: shfl_xor masks 1,2,4,8 stay inside each
        // 16-lane half of the wave32.
#pragma unroll
        for (int r = 0; r < 8; ++r) {
            float v = partial[r];
            v += __shfl_xor(v, 1, 32);
            v += __shfl_xor(v, 2, 32);
            v += __shfl_xor(v, 4, 32);
            v += __shfl_xor(v, 8, 32);
            partial[r] = v;
        }

        if (m == 0) {                     // lanes 0 (rows 0-7) and 16 (rows 8-15)
#pragma unroll
            for (int r = 0; r < 8; ++r) {
                const int row = r + 8 * khalf;
                const float logit = partial[r] + b2v;
                out[tile * 16 + row] = 1.f / (1.f + __expf(-logit));
            }
        }
    }
}

// ---------------------------------------------------------------------------
extern "C" void kernel_launch(void* const* d_in, const int* in_sizes, int n_in,
                              void* d_out, int out_size, void* d_ws, size_t ws_size,
                              hipStream_t stream) {
    const float*     x          = (const float*)d_in[0];
    const long long* edge_index = (const long long*)d_in[1];
    const float*     edge_attr  = (const float*)d_in[2];
    const float*     W_enc      = (const float*)d_in[3];
    const float*     b_enc      = (const float*)d_in[4];
    const float*     W1         = (const float*)d_in[5];
    const float*     b1         = (const float*)d_in[6];
    const float*     W2         = (const float*)d_in[7];
    const float*     b2         = (const float*)d_in[8];
    float*           out        = (float*)d_out;
    _Float16*        x_enc      = (_Float16*)d_ws;     // 100000*32*2 = 6.4 MB

    (void)in_sizes; (void)n_in; (void)out_size; (void)ws_size;

    // Encoder: 6250 tiles over 250 blocks * 8 waves -> ~3 tiles/wave.
    node_encode_kernel<<<250, 256, 0, stream>>>(x, W_enc, b_enc, x_enc);

    // Edge MLP: 200000 tiles over 2500 blocks * 8 waves -> 10 tiles/wave.
    edge_mlp_kernel<<<2500, 256, 0, stream>>>(x_enc, edge_index, edge_attr,
                                              W1, b1, W2, b2, out);
}
